// GCN_18339510354234
// MI455X (gfx1250) — matrix operand order repaired and checked by reading, not compiled, not run on hardware
//
#include <hip/hip_runtime.h>
#include <stdint.h>

typedef __attribute__((ext_vector_type(8)))  float  v8f;
typedef __attribute__((ext_vector_type(16))) __bf16 v16bf;

union AccU   { v8f  v; float    f[8]; };
union BFragU { v16bf v; unsigned u[8]; };

// fp32 -> bf16 pair with round-to-nearest-even, packed into one dword
__device__ __forceinline__ unsigned bpack2(float lo, float hi) {
  unsigned a = __float_as_uint(lo), b = __float_as_uint(hi);
  a = (a + 0x7FFFu + ((a >> 16) & 1u)) >> 16;
  b = (b + 0x7FFFu + ((b >> 16) & 1u)) >> 16;
  return (b << 16) | (a & 0xFFFFu);
}

// ---------------- degree kernels ----------------
__global__ void gcn_count_deg(const int* __restrict__ src, const int* __restrict__ dst,
                              int* __restrict__ cs, int* __restrict__ cd, int E) {
  int i = blockIdx.x * blockDim.x + threadIdx.x;
  if (i < E) {
    atomicAdd(&cs[src[i]], 1);
    atomicAdd(&cd[dst[i]], 1);
  }
}

__global__ void gcn_deg_rsqrt(const int* __restrict__ c, float* __restrict__ dinv, int n) {
  int i = blockIdx.x * blockDim.x + threadIdx.x;
  if (i < n) {
    float v = (float)(c[i] < 1 ? 1 : c[i]);
    dinv[i] = rsqrtf(v);
  }
}

// ---------------- W pack: fp32 row-major [DIN][DOUT] -> WMMA B-fragment bf16 ----
// Fragment (kt, j, lane): 16 bf16, element e -> W[kt*32 + (lane<16?0:8) + (e<8 ? e : e+8)][j*16 + lane%16]
__global__ void gcn_pack_w(const float* __restrict__ W, uint4* __restrict__ Wpk,
                           int DIN, int DOUT) {
  int t = blockIdx.x * blockDim.x + threadIdx.x;
  int T = DOUT >> 4, KT = DIN >> 5;
  int total = KT * T * 32;
  if (t >= total) return;
  int lane = t & 31;
  int j  = (t >> 5) % T;
  int kt = (t >> 5) / T;
  int n = j * 16 + (lane & 15);
  int kbase = kt * 32 + ((lane >> 4) << 3);
  unsigned u[8];
#pragma unroll
  for (int e = 0; e < 8; ++e) {
    float lo = W[(size_t)(kbase + 2 * e + ((e >> 2) << 3)) * DOUT + n];
    float hi = W[(size_t)(kbase + 2 * e + ((e >> 2) << 3) + 1) * DOUT + n];
    u[e] = bpack2(lo, hi);
  }
  uint4* p = Wpk + (size_t)t * 2;
  p[0] = make_uint4(u[0], u[1], u[2], u[3]);
  p[1] = make_uint4(u[4], u[5], u[6], u[7]);
}

// ---------------- edge scatter: agg[dst] += h[src] * deg_out_inv[src] ----------------
// one wave per edge; lane covers D/128 float4 chunks (coalesced 512B gathers)
template <int D>
__global__ void gcn_scatter(const float* __restrict__ h, const float* __restrict__ dov,
                            const int* __restrict__ src, const int* __restrict__ dst,
                            float* __restrict__ agg, int E) {
  int wid  = (blockIdx.x * blockDim.x + threadIdx.x) >> 5;
  int lane = threadIdx.x & 31;
  if (wid >= E) return;
  int s = src[wid], d = dst[wid];
  float w = dov[s];
  const float4* hs = (const float4*)(h + (size_t)s * D);
  float* ad = agg + (size_t)d * D;
#pragma unroll
  for (int i = 0; i < D / 128; ++i) {
    float4 v = hs[lane + i * 32];
    int c = (lane + i * 32) * 4;
    unsafeAtomicAdd(ad + c + 0, v.x * w);
    unsafeAtomicAdd(ad + c + 1, v.y * w);
    unsafeAtomicAdd(ad + c + 2, v.z * w);
    unsafeAtomicAdd(ad + c + 3, v.w * w);
  }
}

// ---------------- fused GEMM + bias + deg_in scale (+ LayerNorm + PReLU) ----------------
// one wave per 16-row tile; accumulates all DOUT columns (T=DOUT/16 WMMA col-tiles)
template <int DIN, int DOUT, bool LN>
__global__ __launch_bounds__(256) void gcn_gemm(
    const float* __restrict__ agg, const uint4* __restrict__ Wpk,
    const float* __restrict__ bias, const float* __restrict__ gamma,
    const float* __restrict__ beta, const float* __restrict__ alpha,
    const float* __restrict__ dinv_in, float* __restrict__ out, int nnodes) {
  constexpr int T  = DOUT / 16;
  constexpr int KT = DIN / 32;
  int wid  = (blockIdx.x * blockDim.x + threadIdx.x) >> 5;
  int lane = threadIdx.x & 31;
  int mt = wid;
  if (mt * 16 >= nnodes) return;   // wave-uniform exit: EXEC stays all-ones for WMMA
  int m16    = lane & 15;
  int half16 = lane >> 4;

  AccU acc[T];
#pragma unroll
  for (int j = 0; j < T; ++j)
#pragma unroll
    for (int w = 0; w < 8; ++w) acc[j].f[w] = 0.0f;

  const float* arow = agg + (size_t)(mt * 16 + m16) * DIN;

  for (int kt = 0; kt < KT; ++kt) {
    // A fragment (16-bit A 16x32 layout): lane<16 -> K {0..7,16..23}, lane>=16 -> +8
    const float* ap = arow + kt * 32 + half16 * 8;
    float4 a0 = *(const float4*)(ap + 0);
    float4 a1 = *(const float4*)(ap + 4);
    float4 a2 = *(const float4*)(ap + 16);
    float4 a3 = *(const float4*)(ap + 20);
    BFragU af;
    af.u[0] = bpack2(a0.x, a0.y); af.u[1] = bpack2(a0.z, a0.w);
    af.u[2] = bpack2(a1.x, a1.y); af.u[3] = bpack2(a1.z, a1.w);
    af.u[4] = bpack2(a2.x, a2.y); af.u[5] = bpack2(a2.z, a2.w);
    af.u[6] = bpack2(a3.x, a3.y); af.u[7] = bpack2(a3.z, a3.w);

    const uint4* bp = Wpk + ((size_t)kt * T * 32 + lane) * 2;
#pragma unroll
    for (int j = 0; j < T; ++j) {
      uint4 b0 = bp[(size_t)j * 64];
      uint4 b1 = bp[(size_t)j * 64 + 1];
      BFragU bf;
      bf.u[0] = b0.x; bf.u[1] = b0.y; bf.u[2] = b0.z; bf.u[3] = b0.w;
      bf.u[4] = b1.x; bf.u[5] = b1.y; bf.u[6] = b1.z; bf.u[7] = b1.w;
      acc[j].v = __builtin_amdgcn_wmma_f32_16x16x32_bf16(
          false, af.v, false, bf.v, (short)0, acc[j].v, false, false);
    }
  }

  // epilogue: C/D layout -> lane holds col = j*16 + m16, VGPR r holds row r + half16*8
  float a_slope = 0.0f;
  if (LN) a_slope = alpha[0];
#pragma unroll
  for (int r = 0; r < 8; ++r) {
    int row = mt * 16 + r + half16 * 8;
    float d = dinv_in[row];
    float xs[T];
    float sum = 0.0f, sq = 0.0f;
#pragma unroll
    for (int j = 0; j < T; ++j) {
      float xv = (acc[j].f[r] + bias[j * 16 + m16]) * d;
      xs[j] = xv;
      sum += xv;
      sq  += xv * xv;
    }
    if (LN) {
      // row lives in one 16-lane half; xor-reduce stays within the half
#pragma unroll
      for (int off = 8; off >= 1; off >>= 1) {
        sum += __shfl_xor(sum, off, 32);
        sq  += __shfl_xor(sq,  off, 32);
      }
      float mu   = sum * (1.0f / DOUT);
      float var  = sq  * (1.0f / DOUT) - mu * mu;
      float rstd = rsqrtf(var + 1e-5f);
#pragma unroll
      for (int j = 0; j < T; ++j) {
        int col = j * 16 + m16;
        float y = gamma[col] * (xs[j] - mu) * rstd + beta[col];
        y = (y >= 0.0f) ? y : a_slope * y;
        out[(size_t)row * DOUT + col] = y;
      }
    } else {
#pragma unroll
      for (int j = 0; j < T; ++j)
        out[(size_t)row * DOUT + j * 16 + m16] = xs[j];
    }
  }
}

extern "C" void kernel_launch(void* const* d_in, const int* in_sizes, int n_in,
                              void* d_out, int out_size, void* d_ws, size_t ws_size,
                              hipStream_t stream) {
  (void)n_in; (void)out_size; (void)ws_size;
  const float* feat = (const float*)d_in[0];
  const float* W1   = (const float*)d_in[1];
  const float* b1   = (const float*)d_in[2];
  const float* g1   = (const float*)d_in[3];
  const float* be1  = (const float*)d_in[4];
  const float* a1   = (const float*)d_in[5];
  const float* W2   = (const float*)d_in[6];
  const float* b2   = (const float*)d_in[7];
  const float* g2   = (const float*)d_in[8];
  const float* be2  = (const float*)d_in[9];
  const float* a2   = (const float*)d_in[10];
  const float* W3   = (const float*)d_in[11];
  const float* b3   = (const float*)d_in[12];
  const int*   src  = (const int*)d_in[13];
  const int*   dst  = (const int*)d_in[14];

  const int Nn = in_sizes[0] / 128;   // nodes
  const int Ee = in_sizes[13];        // edges

  // -------- workspace layout --------
  char* ws = (char*)d_ws;
  size_t off = 0;
  auto take = [&](size_t bytes) -> char* {
    char* p = ws + off;
    off = (off + bytes + 255) & ~(size_t)255;
    return p;
  };
  float* bufA = (float*)take((size_t)Nn * 256 * 4);  // agg buffer
  float* bufB = (float*)take((size_t)Nn * 256 * 4);  // hidden activations
  int*   cnt  = (int*)take((size_t)Nn * 2 * 4);      // [cs | cd] contiguous
  int*   cs   = cnt;
  int*   cd   = cnt + Nn;
  float* dvo  = (float*)take((size_t)Nn * 4);        // deg_out^-1/2
  float* dvi  = (float*)take((size_t)Nn * 4);        // deg_in^-1/2
  uint4* wpk1 = (uint4*)take((size_t)(128 / 32) * (256 / 16) * 32 * 32);
  uint4* wpk2 = (uint4*)take((size_t)(256 / 32) * (256 / 16) * 32 * 32);
  uint4* wpk3 = (uint4*)take((size_t)(256 / 32) * (128 / 16) * 32 * 32);

  const int tpb = 256;
  const int mtiles     = (Nn + 15) / 16;
  const int gemmBlocks = (mtiles + 7) / 8;     // 8 waves / block
  const int scatBlocks = (Ee + 7) / 8;         // 1 wave / edge

  // degrees
  hipMemsetAsync(cnt, 0, (size_t)Nn * 2 * 4, stream);
  gcn_count_deg<<<(Ee + tpb - 1) / tpb, tpb, 0, stream>>>(src, dst, cs, cd, Ee);
  gcn_deg_rsqrt<<<(Nn + tpb - 1) / tpb, tpb, 0, stream>>>(cs, dvo, Nn);
  gcn_deg_rsqrt<<<(Nn + tpb - 1) / tpb, tpb, 0, stream>>>(cd, dvi, Nn);

  // pre-pack weights into WMMA B-fragment bf16 layout (done every launch: deterministic)
  gcn_pack_w<<<(2048 + tpb - 1) / tpb, tpb, 0, stream>>>(W1, wpk1, 128, 256);
  gcn_pack_w<<<(4096 + tpb - 1) / tpb, tpb, 0, stream>>>(W2, wpk2, 256, 256);
  gcn_pack_w<<<(2048 + tpb - 1) / tpb, tpb, 0, stream>>>(W3, wpk3, 256, 128);

  // layer 1: feat[N,128] -> h1[N,256] (LN + PReLU)
  hipMemsetAsync(bufA, 0, (size_t)Nn * 128 * 4, stream);
  gcn_scatter<128><<<scatBlocks, tpb, 0, stream>>>(feat, dvo, src, dst, bufA, Ee);
  gcn_gemm<128, 256, true><<<gemmBlocks, tpb, 0, stream>>>(
      bufA, wpk1, b1, g1, be1, a1, dvi, bufB, Nn);

  // layer 2: h1[N,256] -> h2[N,256] (LN + PReLU)
  hipMemsetAsync(bufA, 0, (size_t)Nn * 256 * 4, stream);
  gcn_scatter<256><<<scatBlocks, tpb, 0, stream>>>(bufB, dvo, src, dst, bufA, Ee);
  gcn_gemm<256, 256, true><<<gemmBlocks, tpb, 0, stream>>>(
      bufA, wpk2, b2, g2, be2, a2, dvi, bufB, Nn);

  // layer 3: h2[N,256] -> out[N,128] (no LN / activation)
  hipMemsetAsync(bufA, 0, (size_t)Nn * 256 * 4, stream);
  gcn_scatter<256><<<scatBlocks, tpb, 0, stream>>>(bufB, dvo, src, dst, bufA, Ee);
  gcn_gemm<256, 128, false><<<gemmBlocks, tpb, 0, stream>>>(
      bufA, wpk3, b3, nullptr, nullptr, nullptr, dvi, (float*)d_out, Nn);
}
